// VAE_62062277427498
// MI455X (gfx1250) — matrix operand order repaired and checked by reading, not compiled, not run on hardware
//
#include <hip/hip_runtime.h>

#define SEQL    2048
#define RIN     512
#define HID     2048
#define G3H     6144     // 3*HID
#define ZL      256
#define PERF_N  384
#define SCORE_N 128
#define DEC_IN  768      // PERF_N + SCORE_N + ZL

#define NB_E 48          // cooperating blocks per encoder direction (1 tile/wave)
#define NB_D 48          // cooperating blocks for decoder
#define GW_D (NB_D * 8)  // decoder global wave count (256-thread blocks)

typedef __attribute__((ext_vector_type(16))) __bf16 v16bf;
typedef __attribute__((ext_vector_type(8)))  __bf16 v8bf;
typedef __attribute__((ext_vector_type(8)))  float  v8f;

// ---------------- helpers ----------------

__device__ __forceinline__ unsigned short f2bf(float f) {
    unsigned u = __float_as_uint(f);
    u += 0x7FFFu + ((u >> 16) & 1u);      // round-to-nearest-even
    return (unsigned short)(u >> 16);
}

__device__ __forceinline__ float sigf(float x) { return 1.0f / (1.0f + __expf(-x)); }

// A-fragment loader for 16-bit 16x32 A tiles: per-lane K = [kb..kb+15] (contiguous)
__device__ __forceinline__ v16bf ld_a16(const unsigned short* p) {
    v8bf lo = *(const v8bf*)p;
    v8bf hi = *(const v8bf*)(p + 8);
    v16bf r;
#pragma unroll
    for (int i = 0; i < 8; i++) { r[i] = lo[i]; r[i + 8] = hi[i]; }
    return r;
}

// One 16-row tile of out = W(rows x K, row-major bf16) @ v (bf16, K elems).
// B operand is v replicated into all 16 columns, so every column of D is the
// matvec; lanes 0 and 16 scatter the 16 row results.
__device__ __forceinline__ void wave_tile_matvec(const unsigned short* __restrict__ W, int K,
                                                 const unsigned short* __restrict__ v,
                                                 float* __restrict__ outv,
                                                 int tile, int lane)
{
    const int rl  = lane & 15;
    const int hlf = lane >> 4;            // 0: K-low half, 1: K-high half
    const unsigned short* Wr = W + (size_t)(tile * 16 + rl) * (size_t)K;
    v8f acc = {};
    for (int kc = 0; kc < K; kc += 32) {
        if (kc + 128 < K) __builtin_prefetch(Wr + kc + 128, 0, 0);
        v16bf a = ld_a16(Wr + kc + hlf * 16);
        v16bf b = *(const v16bf*)(v + kc + hlf * 16);
        acc = __builtin_amdgcn_wmma_f32_16x16x32_bf16(false, a, false, b,
                                                      (short)0, acc, false, false);
    }
    if (rl == 0) {
        const int rbase = tile * 16 + hlf * 8;
#pragma unroll
        for (int i = 0; i < 8; i++) outv[rbase + i] = acc[i];
    }
}

// Device-wide sense-reversing barrier (counter+epoch in global memory).
__device__ __forceinline__ void coop_barrier(unsigned* cnt, unsigned* epoch, unsigned nb) {
    __syncthreads();
    if (threadIdx.x == 0) {
        __threadfence();
        unsigned e    = __hip_atomic_load(epoch, __ATOMIC_ACQUIRE, __HIP_MEMORY_SCOPE_AGENT);
        unsigned prev = __hip_atomic_fetch_add(cnt, 1u, __ATOMIC_ACQ_REL, __HIP_MEMORY_SCOPE_AGENT);
        if (prev == nb - 1u) {
            __hip_atomic_store(cnt, 0u, __ATOMIC_RELAXED, __HIP_MEMORY_SCOPE_AGENT);
            __hip_atomic_fetch_add(epoch, 1u, __ATOMIC_ACQ_REL, __HIP_MEMORY_SCOPE_AGENT);
        } else {
            while (__hip_atomic_load(epoch, __ATOMIC_ACQUIRE, __HIP_MEMORY_SCOPE_AGENT) == e)
                __builtin_amdgcn_s_sleep(1);
        }
    }
    __syncthreads();
}

// ---------------- kernels ----------------

__global__ void init_sync_kernel(unsigned* __restrict__ s) {
    if (threadIdx.x < 16) s[threadIdx.x] = 0u;
}

__global__ void cvt_bf16_kernel(const float* __restrict__ src,
                                unsigned short* __restrict__ dst, int n)
{
    int i = blockIdx.x * 256 + threadIdx.x;
    if (i < n) dst[i] = f2bf(src[i]);
}

// gi[dir][s][m] = wih_dir @ x[time(s)] + bih_dir  (time reversed for dir==1)
// One wave per 16x16 tile. M = 6144 (384 tiles), N = 2048 steps (128 tiles), K = 512.
__global__ __launch_bounds__(256) void gi_gemm_kernel(
    const unsigned short* __restrict__ wfb, const unsigned short* __restrict__ wbb,
    const unsigned short* __restrict__ xb,
    const float* __restrict__ bih_f, const float* __restrict__ bih_b,
    float* __restrict__ gi)
{
    const int gw   = blockIdx.x * 8 + (threadIdx.x >> 5);
    const int lane = threadIdx.x & 31;
    const int TILES = 384 * 128;
    const int dir = gw / TILES;
    const int rem = gw - dir * TILES;
    const int mt  = rem >> 7;
    const int nt  = rem & 127;

    const unsigned short* W   = dir ? wbb : wfb;
    const float*          bih = dir ? bih_b : bih_f;

    const int col  = lane & 15;
    const int hlf  = lane >> 4;
    const int n    = nt * 16 + col;                 // scan-step index
    const int trow = dir ? (SEQL - 1 - n) : n;      // time index into x

    const unsigned short* Wr = W  + (size_t)(mt * 16 + col) * RIN;
    const unsigned short* Bc = xb + (size_t)trow * RIN;

    v8f acc = {};
    for (int kc = 0; kc < RIN; kc += 32) {
        v16bf a = ld_a16(Wr + kc + hlf * 16);
        v16bf b = *(const v16bf*)(Bc + kc + hlf * 16);
        acc = __builtin_amdgcn_wmma_f32_16x16x32_bf16(false, a, false, b,
                                                      (short)0, acc, false, false);
    }
    float*       out = gi + ((size_t)dir * SEQL + n) * G3H + mt * 16 + hlf * 8;
    const float* bp  = bih + mt * 16 + hlf * 8;
#pragma unroll
    for (int i = 0; i < 8; i++) out[i] = acc[i] + bp[i];
}

// Cooperative recurrent encoder: gridDim.x = 2*NB_E, blockIdx.x/NB_E = direction.
// Each of the 384 waves per direction owns one 16-row tile of the 6144x2048
// recurrent matvec (WMMA); gh is double-buffered on step parity in global L2;
// one device barrier per step; gates recomputed redundantly per block into
// block-local LDS hidden state (removes the second barrier).
__global__ __launch_bounds__(256) void encoder_kernel(
    const unsigned short* __restrict__ whh_f_bf, const unsigned short* __restrict__ whh_b_bf,
    const float* __restrict__ gi_all,
    const float* __restrict__ bhh_f, const float* __restrict__ bhh_b,
    float* __restrict__ gh_buf,        // [2 dir][2 parity][G3H]
    unsigned* __restrict__ sync,
    float* __restrict__ h_final)
{
    __shared__ __attribute__((aligned(32))) float          hcur[HID];
    __shared__ __attribute__((aligned(32))) unsigned short hb[HID];

    const int dir = blockIdx.x / NB_E;
    const int b   = blockIdx.x % NB_E;
    const unsigned short* W   = dir ? whh_b_bf : whh_f_bf;
    const float*          bhh = dir ? bhh_b    : bhh_f;
    const float*          gi  = gi_all + (size_t)dir * SEQL * G3H;
    unsigned* cnt = sync + (size_t)dir * 2;
    unsigned* ep  = cnt + 1;

    const int tid  = threadIdx.x;
    const int wave = tid >> 5, lane = tid & 31;
    const int tile = b * 8 + wave;                 // 0..383, one tile per wave

    for (int j = tid; j < HID; j += 256) { hcur[j] = 0.0f; hb[j] = 0; }
    __syncthreads();

    for (int t = 0; t < SEQL; t++) {
        float* ghw = gh_buf + ((size_t)dir * 2 + (t & 1)) * G3H;
        wave_tile_matvec(W, HID, hb, ghw, tile, lane);
        coop_barrier(cnt, ep, NB_E);
        const float* git = gi + (size_t)t * G3H;
        for (int j = tid; j < HID; j += 256) {
            float r = sigf(git[j]            + ghw[j]           + bhh[j]);
            float z = sigf(git[HID + j]      + ghw[HID + j]     + bhh[HID + j]);
            float n = tanhf(git[2*HID + j]   + r * (ghw[2*HID + j] + bhh[2*HID + j]));
            float hn = (1.0f - z) * n + z * hcur[j];
            hcur[j] = hn;
            hb[j]   = f2bf(hn);
        }
        __syncthreads();
    }
    if (b == 0)
        for (int j = tid; j < HID; j += 256) h_final[dir * HID + j] = hcur[j];
}

// Latent head: mean/std to d_out, z (bf16) and h0 to workspace. Tiny; scalar VALU.
__global__ __launch_bounds__(256) void latent_kernel(
    const float* __restrict__ hfinal,
    const float* __restrict__ w_mu, const float* __restrict__ b_mu,
    const float* __restrict__ w_var, const float* __restrict__ b_var,
    const float* __restrict__ w_init, const float* __restrict__ b_init,
    float* __restrict__ mean_out, float* __restrict__ std_out,
    unsigned short* __restrict__ z_bf, float* __restrict__ h0_out)
{
    __shared__ float zsh[ZL];
    const int tid = threadIdx.x;
    if (tid < ZL) {
        const float* wm = w_mu  + (size_t)tid * 2 * HID;
        const float* wv = w_var + (size_t)tid * 2 * HID;
        float sm = 0.0f, sv = 0.0f;
        for (int k = 0; k < 2 * HID; k++) { float h = hfinal[k]; sm += wm[k] * h; sv += wv[k] * h; }
        float mean = sm + b_mu[tid];
        float lv   = sv + b_var[tid];
        mean_out[tid] = mean;
        std_out[tid]  = __expf(0.5f * lv);
        zsh[tid]  = mean;
        z_bf[tid] = f2bf(mean);
    }
    __syncthreads();
    for (int r = tid; r < HID; r += 256) {
        const float* wi = w_init + (size_t)r * ZL;
        float s = 0.0f;
        for (int k = 0; k < ZL; k++) s += wi[k] * zsh[k];
        h0_out[r] = tanhf(s + b_init[r]);
    }
}

// Cooperative decoder: NB_D blocks, 4 device barriers per step (after each
// cell's matvec pair and the output head). Each block keeps its own LDS copy
// of h0/h1/h2/inp and recomputes gates + softmax/argmax redundantly so only
// the WMMA matvec results cross blocks (via three rotating L2 buffer pairs).
__global__ __launch_bounds__(256) void decoder_kernel(
    const unsigned short* __restrict__ c0wi, const unsigned short* __restrict__ c0wh,
    const unsigned short* __restrict__ c1wi, const unsigned short* __restrict__ c1wh,
    const unsigned short* __restrict__ c2wi, const unsigned short* __restrict__ c2wh,
    const unsigned short* __restrict__ wob,
    const float* __restrict__ c0bih, const float* __restrict__ c0bhh,
    const float* __restrict__ c1bih, const float* __restrict__ c1bhh,
    const float* __restrict__ c2bih, const float* __restrict__ c2bhh,
    const float* __restrict__ b_out,
    const float* __restrict__ x,
    const float* __restrict__ h0init, const unsigned short* __restrict__ zbf,
    float* __restrict__ gbuf,          // 6 buffers of G3H floats: g0i g0h g1i g1h g2i g2h
    float* __restrict__ oscg,          // PERF_N floats
    unsigned* __restrict__ sync,
    float* __restrict__ ys)
{
    __shared__ __attribute__((aligned(32))) float          h0s[HID];
    __shared__ __attribute__((aligned(32))) float          h1s[HID];
    __shared__ __attribute__((aligned(32))) float          h2s[HID];
    __shared__ __attribute__((aligned(32))) unsigned short hb0[HID];
    __shared__ __attribute__((aligned(32))) unsigned short hb1[HID];
    __shared__ __attribute__((aligned(32))) unsigned short hb2[HID];
    __shared__ __attribute__((aligned(32))) unsigned short inpb[DEC_IN];
    __shared__ float oscl[PERF_N];
    __shared__ float s_seg[3];
    __shared__ int   s_amax;

    float* g0i = gbuf;             float* g0h = g0i + G3H;
    float* g1i = g0h + G3H;        float* g1h = g1i + G3H;
    float* g2i = g1h + G3H;        float* g2h = g2i + G3H;
    unsigned* cnt = sync + 4;
    unsigned* ep  = cnt + 1;

    const int tid  = threadIdx.x;
    const int wave = tid >> 5, lane = tid & 31;
    const int gw   = blockIdx.x * 8 + wave;        // 0..GW_D-1

    for (int j = tid; j < HID; j += 256) {
        float h = h0init[j];
        h0s[j] = h;    hb0[j] = f2bf(h);
        h1s[j] = 0.0f; hb1[j] = 0;
        h2s[j] = 0.0f; hb2[j] = 0;
    }
    if (tid < PERF_N)  inpb[tid]                    = f2bf(tid == PERF_N - 1  ? 1.0f : 0.0f);
    if (tid < SCORE_N) inpb[PERF_N + tid]           = f2bf(tid == SCORE_N - 1 ? 1.0f : 0.0f);
    if (tid < ZL)      inpb[PERF_N + SCORE_N + tid] = zbf[tid];
    __syncthreads();

    for (int t = 0; t < SCORE_N; t++) {
        // --- cell 0 matvecs: 384 tiles wih@inp, 384 tiles whh@h0 ---
        for (int tl = gw; tl < 768; tl += GW_D) {
            if (tl < 384) wave_tile_matvec(c0wi, DEC_IN, inpb, g0i, tl, lane);
            else          wave_tile_matvec(c0wh, HID,    hb0,  g0h, tl - 384, lane);
        }
        coop_barrier(cnt, ep, NB_D);
        for (int j = tid; j < HID; j += 256) {
            float r = sigf(g0i[j]          + c0bih[j]          + g0h[j]          + c0bhh[j]);
            float z = sigf(g0i[HID + j]    + c0bih[HID + j]    + g0h[HID + j]    + c0bhh[HID + j]);
            float n = tanhf(g0i[2*HID + j] + c0bih[2*HID + j]  + r * (g0h[2*HID + j] + c0bhh[2*HID + j]));
            float hn = (1.0f - z) * n + z * h0s[j];
            h0s[j] = hn; hb0[j] = f2bf(hn);
        }
        __syncthreads();
        // --- cell 1 (hidden = h0n at first step) ---
        const unsigned short* hb1p = (t == 0) ? hb0 : hb1;
        for (int tl = gw; tl < 768; tl += GW_D) {
            if (tl < 384) wave_tile_matvec(c1wi, HID, hb0,  g1i, tl, lane);
            else          wave_tile_matvec(c1wh, HID, hb1p, g1h, tl - 384, lane);
        }
        coop_barrier(cnt, ep, NB_D);
        {
            const float* hp = (t == 0) ? h0s : h1s;
            for (int j = tid; j < HID; j += 256) {
                float r = sigf(g1i[j]          + c1bih[j]          + g1h[j]          + c1bhh[j]);
                float z = sigf(g1i[HID + j]    + c1bih[HID + j]    + g1h[HID + j]    + c1bhh[HID + j]);
                float n = tanhf(g1i[2*HID + j] + c1bih[2*HID + j]  + r * (g1h[2*HID + j] + c1bhh[2*HID + j]));
                float hn = (1.0f - z) * n + z * hp[j];
                h1s[j] = hn; hb1[j] = f2bf(hn);
            }
        }
        __syncthreads();
        // --- cell 2 ---
        const unsigned short* hb2p = (t == 0) ? hb1 : hb2;
        for (int tl = gw; tl < 768; tl += GW_D) {
            if (tl < 384) wave_tile_matvec(c2wi, HID, hb1,  g2i, tl, lane);
            else          wave_tile_matvec(c2wh, HID, hb2p, g2h, tl - 384, lane);
        }
        coop_barrier(cnt, ep, NB_D);
        {
            const float* hp = (t == 0) ? h1s : h2s;
            for (int j = tid; j < HID; j += 256) {
                float r = sigf(g2i[j]          + c2bih[j]          + g2h[j]          + c2bhh[j]);
                float z = sigf(g2i[HID + j]    + c2bih[HID + j]    + g2h[HID + j]    + c2bhh[HID + j]);
                float n = tanhf(g2i[2*HID + j] + c2bih[2*HID + j]  + r * (g2h[2*HID + j] + c2bhh[2*HID + j]));
                float hn = (1.0f - z) * n + z * hp[j];
                h2s[j] = hn; hb2[j] = f2bf(hn);
            }
        }
        __syncthreads();
        // --- output head: 24 tiles ---
        for (int tl = gw; tl < 24; tl += GW_D)
            wave_tile_matvec(wob, HID, hb2, oscg, tl, lane);
        coop_barrier(cnt, ep, NB_D);
        // --- softmax / argmax / feedback (redundant per block) ---
        if (tid < PERF_N) oscl[tid] = oscg[tid] + b_out[tid];
        __syncthreads();
        if (tid < 3) {
            const float* seg = oscl + tid * SCORE_N;
            float mx = seg[0];
            for (int k = 1; k < SCORE_N; k++) mx = fmaxf(mx, seg[k]);
            float s = 0.0f;
            for (int k = 0; k < SCORE_N; k++) s += __expf(seg[k] - mx);
            s_seg[tid] = mx + __logf(s);
        }
        __syncthreads();
        if (tid < PERF_N) {
            float l = oscl[tid] - s_seg[tid >> 7];
            if (blockIdx.x == 0) ys[(size_t)t * PERF_N + tid] = l;
            oscl[tid] = l;
        }
        __syncthreads();
        if (tid == 0) {
            int bi = 0; float bv = oscl[0];
            for (int j = 1; j < PERF_N; j++) { if (oscl[j] > bv) { bv = oscl[j]; bi = j; } }
            s_amax = bi;
        }
        __syncthreads();
        if (tid < PERF_N)  inpb[tid]          = f2bf(tid == s_amax ? 1.0f : 0.0f);
        if (tid < SCORE_N) inpb[PERF_N + tid] = f2bf(x[(size_t)t * RIN + tid]);
        __syncthreads();
    }
}

// ---------------- launch ----------------

extern "C" void kernel_launch(void* const* d_in, const int* in_sizes, int n_in,
                              void* d_out, int out_size, void* d_ws, size_t ws_size,
                              hipStream_t stream)
{
    (void)in_sizes; (void)n_in; (void)out_size; (void)ws_size;

    const float* x      = (const float*)d_in[0];
    const float* wih_f  = (const float*)d_in[1];
    const float* whh_f  = (const float*)d_in[2];
    const float* bih_f  = (const float*)d_in[3];
    const float* bhh_f  = (const float*)d_in[4];
    const float* wih_b  = (const float*)d_in[5];
    const float* whh_b  = (const float*)d_in[6];
    const float* bih_b  = (const float*)d_in[7];
    const float* bhh_b  = (const float*)d_in[8];
    const float* w_mu   = (const float*)d_in[9];
    const float* b_mu   = (const float*)d_in[10];
    const float* w_var  = (const float*)d_in[11];
    const float* b_var  = (const float*)d_in[12];
    const float* w_init = (const float*)d_in[13];
    const float* b_init = (const float*)d_in[14];
    const float* c0_wih = (const float*)d_in[15];
    const float* c0_whh = (const float*)d_in[16];
    const float* c0_bih = (const float*)d_in[17];
    const float* c0_bhh = (const float*)d_in[18];
    const float* c1_wih = (const float*)d_in[19];
    const float* c1_whh = (const float*)d_in[20];
    const float* c1_bih = (const float*)d_in[21];
    const float* c1_bhh = (const float*)d_in[22];
    const float* c2_wih = (const float*)d_in[23];
    const float* c2_whh = (const float*)d_in[24];
    const float* c2_bih = (const float*)d_in[25];
    const float* c2_bhh = (const float*)d_in[26];
    const float* w_out  = (const float*)d_in[27];
    const float* b_out  = (const float*)d_in[28];

    float* out  = (float*)d_out;
    float* ys   = out;
    float* mean = out + (size_t)SCORE_N * PERF_N;   // 49152
    float* stdo = mean + ZL;

    char*  base = (char*)d_ws;
    size_t off  = 0;
    auto alloc = [&](size_t bytes) -> void* {
        void* p = base + off;
        off = (off + bytes + 255) & ~(size_t)255;
        return p;
    };
    unsigned short* xb   = (unsigned short*)alloc((size_t)SEQL * RIN * 2);
    unsigned short* wfb  = (unsigned short*)alloc((size_t)G3H * RIN * 2);
    unsigned short* wbb  = (unsigned short*)alloc((size_t)G3H * RIN * 2);
    unsigned short* hfwb = (unsigned short*)alloc((size_t)G3H * HID * 2);
    unsigned short* hbwb = (unsigned short*)alloc((size_t)G3H * HID * 2);
    unsigned short* c0wi = (unsigned short*)alloc((size_t)G3H * DEC_IN * 2);
    unsigned short* c1wi = (unsigned short*)alloc((size_t)G3H * HID * 2);
    unsigned short* c2wi = (unsigned short*)alloc((size_t)G3H * HID * 2);
    unsigned short* c0wh = (unsigned short*)alloc((size_t)G3H * HID * 2);
    unsigned short* c1wh = (unsigned short*)alloc((size_t)G3H * HID * 2);
    unsigned short* c2wh = (unsigned short*)alloc((size_t)G3H * HID * 2);
    unsigned short* wob  = (unsigned short*)alloc((size_t)PERF_N * HID * 2);
    float*          gi   = (float*)alloc((size_t)2 * SEQL * G3H * 4);
    float*          hfin = (float*)alloc((size_t)2 * HID * 4);
    unsigned short* zbf  = (unsigned short*)alloc((size_t)ZL * 2);
    float*          h0ws = (float*)alloc((size_t)HID * 4);
    float*          ghb  = (float*)alloc((size_t)2 * 2 * G3H * 4);   // encoder gh, dbl-buffered
    float*          dgb  = (float*)alloc((size_t)6 * G3H * 4);       // decoder matvec buffers
    float*          oscg = (float*)alloc((size_t)PERF_N * 4);
    unsigned*       sync = (unsigned*)alloc(64 * 4);

    init_sync_kernel<<<1, 32, 0, stream>>>(sync);

    auto cvt = [&](const float* s, unsigned short* d, size_t n) {
        int blocks = (int)((n + 255) / 256);
        cvt_bf16_kernel<<<blocks, 256, 0, stream>>>(s, d, (int)n);
    };
    cvt(x,      xb,   (size_t)SEQL * RIN);
    cvt(wih_f,  wfb,  (size_t)G3H * RIN);
    cvt(wih_b,  wbb,  (size_t)G3H * RIN);
    cvt(whh_f,  hfwb, (size_t)G3H * HID);
    cvt(whh_b,  hbwb, (size_t)G3H * HID);
    cvt(c0_wih, c0wi, (size_t)G3H * DEC_IN);
    cvt(c1_wih, c1wi, (size_t)G3H * HID);
    cvt(c2_wih, c2wi, (size_t)G3H * HID);
    cvt(c0_whh, c0wh, (size_t)G3H * HID);
    cvt(c1_whh, c1wh, (size_t)G3H * HID);
    cvt(c2_whh, c2wh, (size_t)G3H * HID);
    cvt(w_out,  wob,  (size_t)PERF_N * HID);

    // gi = wih @ x (+bih), both directions: 2 * 384 * 128 tiles, 8 waves/block
    gi_gemm_kernel<<<(2 * 384 * 128) / 8, 256, 0, stream>>>(wfb, wbb, xb, bih_f, bih_b, gi);

    // cooperative recurrent encoder: 48 blocks per direction, 1 barrier/step
    encoder_kernel<<<2 * NB_E, 256, 0, stream>>>(hfwb, hbwb, gi, bhh_f, bhh_b,
                                                 ghb, sync, hfin);

    // latent head
    latent_kernel<<<1, 256, 0, stream>>>(hfin, w_mu, b_mu, w_var, b_var,
                                         w_init, b_init, mean, stdo, zbf, h0ws);

    // cooperative decoder: 48 blocks, 4 barriers/step
    decoder_kernel<<<NB_D, 256, 0, stream>>>(
        c0wi, c0wh, c1wi, c1wh, c2wi, c2wh, wob,
        c0_bih, c0_bhh, c1_bih, c1_bhh, c2_bih, c2_bhh, b_out,
        x, h0ws, zbf, dgb, oscg, sync, ys);
}
